// MaddnessPreActBlock_62904091018025
// MI455X (gfx1250) — matrix operand order repaired and checked by reading, not compiled, not run on hardware
//
#include <hip/hip_runtime.h>

typedef __attribute__((ext_vector_type(16))) _Float16 v16h;
typedef __attribute__((ext_vector_type(8)))  float    v8f;

#define R_TOT   25088      // 32 * 28 * 28
#define HW_OUT  784        // 28*28
#define P_OUT   256

// ---------------------------------------------------------------------------
// BN stats: NCHW layout (for x). One block per channel.
// ---------------------------------------------------------------------------
__global__ __launch_bounds__(256)
void bn_stats_nchw(const float* __restrict__ x, float* __restrict__ stat,
                   int C, int HW, int N) {
    __shared__ float s1[256];
    __shared__ float s2[256];
    int ch = blockIdx.x, tid = threadIdx.x;
    float a = 0.f, b = 0.f;
    for (int n = 0; n < N; ++n) {
        const float* p = x + ((size_t)n * C + ch) * HW;
        for (int i = tid; i < HW; i += 256) { float v = p[i]; a += v; b += v * v; }
    }
    s1[tid] = a; s2[tid] = b; __syncthreads();
    for (int st = 128; st > 0; st >>= 1) {
        if (tid < st) { s1[tid] += s1[tid + st]; s2[tid] += s2[tid + st]; }
        __syncthreads();
    }
    if (tid == 0) {
        float cnt  = (float)N * (float)HW;
        float mean = s1[0] / cnt;
        float var  = s2[0] / cnt - mean * mean;
        stat[2 * ch]     = mean;
        stat[2 * ch + 1] = rsqrtf(var + 1e-5f);
    }
}

// BN stats: row-major [R][C] layout (for conv1 output). One block per channel.
__global__ __launch_bounds__(256)
void bn_stats_rc(const float* __restrict__ src, float* __restrict__ stat,
                 int C, int R) {
    __shared__ float s1[256];
    __shared__ float s2[256];
    int ch = blockIdx.x, tid = threadIdx.x;
    float a = 0.f, b = 0.f;
    for (int r = tid; r < R; r += 256) {
        float v = src[(size_t)r * C + ch]; a += v; b += v * v;
    }
    s1[tid] = a; s2[tid] = b; __syncthreads();
    for (int st = 128; st > 0; st >>= 1) {
        if (tid < st) { s1[tid] += s1[tid + st]; s2[tid] += s2[tid + st]; }
        __syncthreads();
    }
    if (tid == 0) {
        float cnt  = (float)R;
        float mean = s1[0] / cnt;
        float var  = s2[0] / cnt - mean * mean;
        stat[2 * ch]     = mean;
        stat[2 * ch + 1] = rsqrtf(var + 1e-5f);
    }
}

// ---------------------------------------------------------------------------
// LUT f32 [Cb][16][256] -> transposed f16 [Cb][256][16] (contiguous B frags)
// ---------------------------------------------------------------------------
__global__ __launch_bounds__(256)
void lut_to_f16(const float* __restrict__ lut, _Float16* __restrict__ lutT, int total) {
    int idx = blockIdx.x * 256 + threadIdx.x;
    if (idx >= total) return;
    int o = idx & 255;
    int e = (idx >> 8) & 15;
    int c = idx >> 12;
    lutT[((size_t)c * 256 + o) * 16 + e] = (_Float16)lut[idx];
}

// ---------------------------------------------------------------------------
// Maddness hash-tree encoders. Codes stored [Cb][R] (GEMM-friendly).
// Only the 4 split-dim values are loaded (not the full patch).
// ---------------------------------------------------------------------------
__global__ __launch_bounds__(256)
void encode_conv1(const float* __restrict__ x, const float* __restrict__ stat,
                  const float* __restrict__ g, const float* __restrict__ b,
                  const float* __restrict__ thr, const int* __restrict__ sd,
                  unsigned char* __restrict__ codes) {
    int gid = blockIdx.x * 256 + threadIdx.x;       // over Cb1 * R
    int c = gid / R_TOT;
    int r = gid - c * R_TOT;
    int n = r / HW_OUT; int rem = r - n * HW_OUT;
    int ho = rem / 28, wo = rem - (rem / 28) * 28;
    float mean = stat[2 * c], rstd = stat[2 * c + 1];
    float sc = rstd * g[c];
    float sh = b[c] - mean * sc;
    const float* xc = x + ((size_t)n * 128 + c) * 3136;   // 56*56 plane
    int bucket = 0;
#pragma unroll
    for (int t = 0; t < 4; ++t) {
        int d  = sd[c * 4 + t];                 // 0..8 -> (kh,kw)
        int kh = d / 3, kw = d - kh * 3;
        int h = 2 * ho - 1 + kh, w = 2 * wo - 1 + kw;
        float v = 0.f;
        if ((unsigned)h < 56u && (unsigned)w < 56u)
            v = fmaxf(xc[h * 56 + w] * sc + sh, 0.f);
        int node = (1 << t) - 1 + bucket;
        bucket = 2 * bucket + (v > thr[c * 15 + node] ? 1 : 0);
    }
    codes[(size_t)c * R_TOT + r] = (unsigned char)bucket;
}

__global__ __launch_bounds__(256)
void encode_shortcut(const float* __restrict__ x, const float* __restrict__ stat,
                     const float* __restrict__ g, const float* __restrict__ b,
                     const float* __restrict__ thr, const int* __restrict__ sd,
                     unsigned char* __restrict__ codes) {
    int gid = blockIdx.x * 256 + threadIdx.x;       // over 32 * R
    int c = gid / R_TOT;                            // 0..31
    int r = gid - c * R_TOT;
    int n = r / HW_OUT; int rem = r - n * HW_OUT;
    int ho = rem / 28, wo = rem - (rem / 28) * 28;
    int bucket = 0;
#pragma unroll
    for (int t = 0; t < 4; ++t) {
        int d  = sd[c * 4 + t];                 // 0..3 -> channel 4c+d
        int cc = 4 * c + d;
        float mean = stat[2 * cc], rstd = stat[2 * cc + 1];
        float xv = x[(((size_t)n * 128 + cc) * 3136) + (2 * ho) * 56 + 2 * wo];
        float v = fmaxf((xv - mean) * rstd * g[cc] + b[cc], 0.f);
        int node = (1 << t) - 1 + bucket;
        bucket = 2 * bucket + (v > thr[c * 15 + node] ? 1 : 0);
    }
    codes[(size_t)c * R_TOT + r] = (unsigned char)bucket;
}

__global__ __launch_bounds__(256)
void encode_conv2(const float* __restrict__ src /*[R][256]*/, const float* __restrict__ stat,
                  const float* __restrict__ g, const float* __restrict__ b,
                  const float* __restrict__ thr, const int* __restrict__ sd,
                  unsigned char* __restrict__ codes) {
    int gid = blockIdx.x * 256 + threadIdx.x;       // over 256 * R
    int c = gid / R_TOT;
    int r = gid - c * R_TOT;
    int n = r / HW_OUT; int rem = r - n * HW_OUT;
    int ho = rem / 28, wo = rem - (rem / 28) * 28;
    float mean = stat[2 * c], rstd = stat[2 * c + 1];
    float sc = rstd * g[c];
    float sh = b[c] - mean * sc;
    int bucket = 0;
#pragma unroll
    for (int t = 0; t < 4; ++t) {
        int d  = sd[c * 4 + t];
        int kh = d / 3, kw = d - kh * 3;
        int h = ho - 1 + kh, w = wo - 1 + kw;      // stride 1, pad 1
        float v = 0.f;
        if ((unsigned)h < 28u && (unsigned)w < 28u) {
            float xv = src[((size_t)n * HW_OUT + h * 28 + w) * P_OUT + c];
            v = fmaxf(xv * sc + sh, 0.f);
        }
        int node = (1 << t) - 1 + bucket;
        bucket = 2 * bucket + (v > thr[c * 15 + node] ? 1 : 0);
    }
    codes[(size_t)c * R_TOT + r] = (unsigned char)bucket;
}

// ---------------------------------------------------------------------------
// LUT GEMM via WMMA: out[R][256] = OneHot(codes) [R x 16Cb] @ lutT [16Cb x 256]
// Block = 256 threads (8 waves), covers a 32-row x 256-col output tile:
//   waves 0-3 -> row tile 0, waves 4-7 -> row tile 1
//   wave handles 16 rows x 64 cols: 4 f32 accumulators, 4 WMMA per A build
// (amortizes the one-hot A construction 4x and doubles per-block B reuse).
// A (one-hot) built in registers in contiguous k order:
//   lane group g, element e  <->  K = 16g + e
// B loaded contiguously from lutT[c+g][col][0..15] in the same contiguous k order:
//   lane group g holds K = 16g..16g+15 for column lane&15.
// ---------------------------------------------------------------------------
__global__ __launch_bounds__(256)
void lut_gemm(const unsigned char* __restrict__ codes /*[Cb][R]*/,
              const _Float16* __restrict__ lutT /*[Cb][256][16]*/,
              float* __restrict__ out /*[R][256]*/, int Cb, int R) {
    int tid  = threadIdx.x;
    int lane = tid & 31;
    int wv   = tid >> 5;          // 0..7
    int g    = (lane >> 4) & 1;
    int m    = lane & 15;
    int rt   = wv >> 2;           // row tile within block (0/1)
    int row  = blockIdx.x * 32 + rt * 16 + m;
    int col0 = (wv & 3) * 64 + m; // 64-col slice per wave

    v8f acc0 = {0.f, 0.f, 0.f, 0.f, 0.f, 0.f, 0.f, 0.f};
    v8f acc1 = acc0, acc2 = acc0, acc3 = acc0;

    for (int c = 0; c < Cb; c += 2) {
        int k0 = (int)codes[(size_t)c * R + row];            // 0..15
        int k1 = 16 + (int)codes[(size_t)(c + 1) * R + row]; // 16..31
        __builtin_prefetch((const void*)(codes + (size_t)(c + 2) * R + row), 0, 3);

        v16h a;
#pragma unroll
        for (int e = 0; e < 16; ++e) {
            int ke = e + 16 * g;
            a[e] = (ke == k0 || ke == k1) ? (_Float16)1.0f : (_Float16)0.0f;
        }

        const v16h* bp = (const v16h*)(lutT + (((size_t)(c + g)) * 256 + col0) * 16);
        v16h b0 = bp[0];     // columns col0
        v16h b1 = bp[16];    // columns col0+16
        v16h b2 = bp[32];    // columns col0+32
        v16h b3 = bp[48];    // columns col0+48

        acc0 = __builtin_amdgcn_wmma_f32_16x16x32_f16(false, a, false, b0,
                                                      (short)0, acc0, false, false);
        acc1 = __builtin_amdgcn_wmma_f32_16x16x32_f16(false, a, false, b1,
                                                      (short)0, acc1, false, false);
        acc2 = __builtin_amdgcn_wmma_f32_16x16x32_f16(false, a, false, b2,
                                                      (short)0, acc2, false, false);
        acc3 = __builtin_amdgcn_wmma_f32_16x16x32_f16(false, a, false, b3,
                                                      (short)0, acc3, false, false);
    }

    // D layout: lanes 0-15 VGPR j -> M=j ; lanes 16-31 -> M=8+j ; N = lane&15
    int orow = blockIdx.x * 32 + rt * 16 + 8 * g;
#pragma unroll
    for (int j = 0; j < 8; ++j) {
        float* op = out + (size_t)(orow + j) * P_OUT + col0;
        op[0]  = acc0[j];
        op[16] = acc1[j];
        op[32] = acc2[j];
        op[48] = acc3[j];
    }
}

// ---------------------------------------------------------------------------
// SE gate: avgpool -> fc1(16) -> relu -> fc2(256) -> sigmoid. One block per n.
// ---------------------------------------------------------------------------
__global__ __launch_bounds__(256)
void se_gate(const float* __restrict__ out2, const float* __restrict__ fc1w,
             const float* __restrict__ fc1b, const float* __restrict__ fc2w,
             const float* __restrict__ fc2b, float* __restrict__ gate) {
    __shared__ float pl[256];
    __shared__ float hh[16];
    int n = blockIdx.x, ch = threadIdx.x;
    float s = 0.f;
    const float* p = out2 + (size_t)n * HW_OUT * P_OUT + ch;
    for (int i = 0; i < HW_OUT; ++i) s += p[(size_t)i * P_OUT];
    pl[ch] = s * (1.0f / (float)HW_OUT);
    __syncthreads();
    if (ch < 16) {
        float h = fc1b[ch];
        for (int k = 0; k < 256; ++k) h += fc1w[ch * 256 + k] * pl[k];
        hh[ch] = fmaxf(h, 0.f);
    }
    __syncthreads();
    float v = fc2b[ch];
#pragma unroll
    for (int j = 0; j < 16; ++j) v += fc2w[ch * 16 + j] * hh[j];
    gate[n * 256 + ch] = 1.f / (1.f + expf(-v));
}

// ---------------------------------------------------------------------------
// Fused epilogue: y[n,ch,h,w] = out2[r][ch]*gate[n][ch] + shortcut[r][ch]
// ---------------------------------------------------------------------------
__global__ __launch_bounds__(256)
void fuse_out(const float* __restrict__ out2, const float* __restrict__ sc,
              const float* __restrict__ gate, float* __restrict__ y) {
    size_t idx = (size_t)blockIdx.x * 256 + threadIdx.x; // over 32*256*784
    int hw = (int)(idx % HW_OUT);
    int ch = (int)((idx / HW_OUT) % P_OUT);
    int n  = (int)(idx / ((size_t)HW_OUT * P_OUT));
    size_t r = (size_t)n * HW_OUT + hw;
    y[idx] = out2[r * P_OUT + ch] * gate[n * 256 + ch] + sc[r * P_OUT + ch];
}

// ---------------------------------------------------------------------------
extern "C" void kernel_launch(void* const* d_in, const int* in_sizes, int n_in,
                              void* d_out, int out_size, void* d_ws, size_t ws_size,
                              hipStream_t stream) {
    const float* x     = (const float*)d_in[0];
    const float* bn1g  = (const float*)d_in[1];
    const float* bn1b  = (const float*)d_in[2];
    const float* bn2g  = (const float*)d_in[3];
    const float* bn2b  = (const float*)d_in[4];
    const float* thr1  = (const float*)d_in[5];
    const float* lut1  = (const float*)d_in[6];
    const float* thr2  = (const float*)d_in[7];
    const float* lut2  = (const float*)d_in[8];
    const float* thrs  = (const float*)d_in[9];
    const float* luts  = (const float*)d_in[10];
    const float* fc1w  = (const float*)d_in[11];
    const float* fc1b  = (const float*)d_in[12];
    const float* fc2w  = (const float*)d_in[13];
    const float* fc2b  = (const float*)d_in[14];
    const int*   sd1   = (const int*)d_in[15];
    const int*   sd2   = (const int*)d_in[16];
    const int*   sds   = (const int*)d_in[17];

    const int R = R_TOT;

    // Deterministic workspace carve-up (256B aligned).
    char* ws = (char*)d_ws;
    size_t off = 0;
    auto alloc = [&](size_t bytes) -> void* {
        void* p = ws + off;
        off = (off + bytes + 255) & ~(size_t)255;
        return p;
    };
    float*         stat1  = (float*)alloc(128 * 2 * sizeof(float));
    float*         stat2  = (float*)alloc(256 * 2 * sizeof(float));
    float*         gate   = (float*)alloc(32 * 256 * sizeof(float));
    _Float16*      lut1T  = (_Float16*)alloc((size_t)128 * 256 * 16 * 2);
    _Float16*      lut2T  = (_Float16*)alloc((size_t)256 * 256 * 16 * 2);
    _Float16*      lutsT  = (_Float16*)alloc((size_t)32 * 256 * 16 * 2);
    unsigned char* codes1 = (unsigned char*)alloc((size_t)128 * R + 2 * R); // +2R: prefetch slack
    unsigned char* codes2 = (unsigned char*)alloc((size_t)256 * R + 2 * R);
    unsigned char* codess = (unsigned char*)alloc((size_t)32 * R + 2 * R);
    float*         out1   = (float*)alloc((size_t)R * 256 * sizeof(float));
    float*         out2   = (float*)alloc((size_t)R * 256 * sizeof(float));
    float*         scb    = (float*)alloc((size_t)R * 256 * sizeof(float));
    (void)off; (void)ws_size; (void)in_sizes; (void)n_in; (void)out_size;

    // 1) BN1 stats over x (128 channels of 32*56*56)
    bn_stats_nchw<<<128, 256, 0, stream>>>(x, stat1, 128, 3136, 32);

    // 2) LUT -> transposed f16 fragments
    lut_to_f16<<<2048, 256, 0, stream>>>(lut1, lut1T, 128 * 4096);
    lut_to_f16<<<4096, 256, 0, stream>>>(lut2, lut2T, 256 * 4096);
    lut_to_f16<<< 512, 256, 0, stream>>>(luts, lutsT, 32 * 4096);

    // 3) Encode conv1 (3x3 s2 p1) and shortcut (1x1 s2), BN applied inline
    encode_conv1<<<(128 * R) / 256, 256, 0, stream>>>(x, stat1, bn1g, bn1b, thr1, sd1, codes1);
    encode_shortcut<<<(32 * R) / 256, 256, 0, stream>>>(x, stat1, bn1g, bn1b, thrs, sds, codess);

    // 4) WMMA LUT-GEMMs: conv1 and shortcut (32-row tiles per block)
    lut_gemm<<<R / 32, 256, 0, stream>>>(codes1, lut1T, out1, 128, R);
    lut_gemm<<<R / 32, 256, 0, stream>>>(codess, lutsT, scb, 32, R);

    // 5) BN2 stats over conv1 output, encode conv2 (3x3 s1 p1), GEMM
    bn_stats_rc<<<256, 256, 0, stream>>>(out1, stat2, 256, R);
    encode_conv2<<<(256 * R) / 256, 256, 0, stream>>>(out1, stat2, bn2g, bn2b, thr2, sd2, codes2);
    lut_gemm<<<R / 32, 256, 0, stream>>>(codes2, lut2T, out2, 256, R);

    // 6) SE gate + fused epilogue into NCHW d_out
    se_gate<<<32, 256, 0, stream>>>(out2, fc1w, fc1b, fc2w, fc2b, gate);
    fuse_out<<<(32 * 256 * HW_OUT) / 256, 256, 0, stream>>>(out2, scb, gate, (float*)d_out);
}
